// VQMovingAvg_7275674599498
// MI455X (gfx1250) — matrix operand order repaired and checked by reading, not compiled, not run on hardware
//
#include <hip/hip_runtime.h>
#include <hip/hip_bf16.h>

typedef __attribute__((ext_vector_type(16))) __bf16 v16bf;
typedef __attribute__((ext_vector_type(8)))  float  v8f;
typedef __attribute__((ext_vector_type(4)))  __bf16 v4bf;

#define DIM    512
#define NCODE  1024
#define NTOK   16384            // 8 * 2048
#define QELEMS (NTOK * DIM)     // 8388608

// ---- workspace layout (bytes), all offsets 16B-aligned ----
#define WS_DW     0u            // u64[NCODE*DIM]   4 MiB  (fixed-point dw accumulator)
#define WS_LOSS   4194304u      // u64[1]                   (fixed-point loss accumulator)
#define WS_CNT    4194560u      // int[NCODE]
#define WS_IDX    4198656u      // int[NTOK]
#define WS_CSQ    4264192u      // float[NCODE]  ||c||^2
#define WS_CBNEW  4268288u      // float[NCODE*DIM] updated codebook
#define WS_CBHI   6365440u      // bf16[NCODE*DIM]
#define WS_CBLO   7414016u      // bf16[NCODE*DIM]
#define WS_ZERO_BYTES 4198656u  // zero dw + loss + counts

#define FIXSCALE     16777216.0f        // 2^24
#define INV_FIXSCALE 5.9604644775390625e-08  // 2^-24

// ---------- prep: split codebook into bf16 hi/lo ----------
__global__ void k_prep_cb(const float* __restrict__ cb,
                          __bf16* __restrict__ hi, __bf16* __restrict__ lo) {
    int i = blockIdx.x * 256 + threadIdx.x;     // < NCODE*DIM
    float c = cb[i];
    __bf16 h = (__bf16)c;
    hi[i] = h;
    lo[i] = (__bf16)(c - (float)h);
}

// ---------- prep: ||c||^2 per code ----------
__global__ void k_csq(const float* __restrict__ cb, float* __restrict__ csq) {
    int n = blockIdx.x * 256 + threadIdx.x;     // < NCODE
    const float4* row = (const float4*)(cb + n * DIM);
    float s = 0.f;
    #pragma unroll 8
    for (int i = 0; i < DIM / 4; ++i) {
        float4 v = row[i];
        s += v.x * v.x + v.y * v.y + v.z * v.z + v.w * v.w;
    }
    csq[n] = s;
}

// ---------- phase 1: WMMA distance GEMM + argmin ----------
// block = 64 threads (2 waves); each wave owns 16 tokens; sweep all 64 N-tiles.
#define TPB_TOK 32
__global__ __launch_bounds__(64)
void k_argmin(const float* __restrict__ x,
              const __bf16* __restrict__ cbhi,
              const __bf16* __restrict__ cblo,
              const float*  __restrict__ csq,
              int*   __restrict__ idx_i,
              float* __restrict__ idx_f) {
    extern __shared__ __bf16 smem[];            // [32*512] hi then [32*512] lo = 64 KiB
    __bf16* sh_hi = smem;
    __bf16* sh_lo = smem + TPB_TOK * DIM;

    const int tid  = threadIdx.x;
    const int tok0 = blockIdx.x * TPB_TOK;

    // stage x tile into LDS as bf16 hi + residual lo
    const float4* xv = (const float4*)(x + tok0 * DIM);
    for (int i = tid; i < TPB_TOK * DIM / 4; i += 64) {
        float4 v = xv[i];
        __bf16 h0 = (__bf16)v.x, h1 = (__bf16)v.y, h2 = (__bf16)v.z, h3 = (__bf16)v.w;
        v4bf hv = {h0, h1, h2, h3};
        v4bf lv = {(__bf16)(v.x - (float)h0), (__bf16)(v.y - (float)h1),
                   (__bf16)(v.z - (float)h2), (__bf16)(v.w - (float)h3)};
        *(v4bf*)(sh_hi + i * 4) = hv;
        *(v4bf*)(sh_lo + i * 4) = lv;
    }
    __syncthreads();

    const int lane  = tid & 31;
    const int wv    = tid >> 5;                 // 0..1 : wave's 16-token slice
    const int lm    = lane & 15;
    const int lhalf = lane >> 4;                // 0/1
    const int arow  = (wv * 16 + lm) * DIM;     // A row base (LDS elements)
    const int aoff  = lhalf * 16;               // A: lanes 0-15 K 0..15, lanes 16-31 K 16..31
    const int boff  = lhalf * 16;               // B: lanes 0-15 K 0..15, lanes 16-31 K 16..31

    float best[8];
    int   bidx[8];
    #pragma unroll
    for (int r = 0; r < 8; ++r) { best[r] = 3.4e38f; bidx[r] = 0; }

    union V16 { uint4 u[2]; v16bf v; };

    for (int jt = 0; jt < NCODE / 16; jt += 4) {        // 4 N-tiles per pass
        v8f acc[4] = {};
        const __bf16* browh[4];
        const __bf16* browl[4];
        #pragma unroll
        for (int t = 0; t < 4; ++t) {
            int n = (jt + t) * 16 + lm;
            browh[t] = cbhi + n * DIM;
            browl[t] = cblo + n * DIM;
        }
        if (jt + 4 < NCODE / 16) {                      // hint next tile into cache
            __builtin_prefetch(cbhi + ((jt + 4) * 16 + lm) * DIM, 0, 1);
            __builtin_prefetch(cblo + ((jt + 4) * 16 + lm) * DIM, 0, 1);
        }

        #pragma unroll 4
        for (int kk = 0; kk < DIM / 32; ++kk) {
            const int ab = arow + kk * 32 + aoff;
            V16 ah, al;
            ah.u[0] = *(const uint4*)(sh_hi + ab);
            ah.u[1] = *(const uint4*)(sh_hi + ab + 8);
            al.u[0] = *(const uint4*)(sh_lo + ab);
            al.u[1] = *(const uint4*)(sh_lo + ab + 8);
            #pragma unroll
            for (int t = 0; t < 4; ++t) {
                const __bf16* bph = browh[t] + kk * 32 + boff;
                const __bf16* bpl = browl[t] + kk * 32 + boff;
                V16 bh, bl;
                bh.u[0] = ((const uint4*)bph)[0];
                bh.u[1] = ((const uint4*)bph)[1];
                bl.u[0] = ((const uint4*)bpl)[0];
                bl.u[1] = ((const uint4*)bpl)[1];
                acc[t] = __builtin_amdgcn_wmma_f32_16x16x32_bf16(
                            false, ah.v, false, bh.v, (short)0, acc[t], false, false);
                acc[t] = __builtin_amdgcn_wmma_f32_16x16x32_bf16(
                            false, al.v, false, bh.v, (short)0, acc[t], false, false);
                acc[t] = __builtin_amdgcn_wmma_f32_16x16x32_bf16(
                            false, ah.v, false, bl.v, (short)0, acc[t], false, false);
            }
        }

        // score = ||c||^2 - 2 x.c   (x^2 term constant per token -> dropped)
        #pragma unroll
        for (int t = 0; t < 4; ++t) {
            const int n  = (jt + t) * 16 + lm;
            const float cs = csq[n];
            #pragma unroll
            for (int r = 0; r < 8; ++r) {
                float score = cs - 2.0f * acc[t][r];
                bool take = score < best[r];
                best[r] = take ? score : best[r];
                bidx[r] = take ? n : bidx[r];
            }
        }
    }

    // min-reduce across the 16 lanes of each half (same M rows, different N columns)
    #pragma unroll
    for (int r = 0; r < 8; ++r) {
        float b = best[r]; int bi = bidx[r];
        #pragma unroll
        for (int s = 1; s < 16; s <<= 1) {
            float ob = __shfl_xor(b, s, 32);
            int   oi = __shfl_xor(bi, s, 32);
            bool take = (ob < b) || (ob == b && oi < bi);
            b  = take ? ob : b;
            bi = take ? oi : bi;
        }
        if (lm == r) {   // lanes 0-15 hold M = r, lanes 16-31 hold M = 8 + r
            int token = tok0 + wv * 16 + lhalf * 8 + r;
            idx_i[token] = bi;
            idx_f[token] = (float)bi;
        }
    }
}

// ---------- phase 2: histogram (deterministic integer atomics) ----------
__global__ void k_counts(const int* __restrict__ idx, int* __restrict__ cnt) {
    int t = blockIdx.x * 256 + threadIdx.x;     // < NTOK
    atomicAdd(&cnt[idx[t]], 1);
}

// ---------- phase 2: dw scatter (deterministic i64 fixed-point atomics) ----------
__global__ void k_scatter(const float* __restrict__ x, const int* __restrict__ idx,
                          unsigned long long* __restrict__ dw) {
    int gid   = blockIdx.x * 256 + threadIdx.x; // < NTOK*DIM/4
    int token = gid >> 7;                       // 128 float4 chunks per token
    int chunk = gid & 127;
    float4 v = ((const float4*)x)[gid];
    int n = idx[token];
    unsigned long long* base = dw + (size_t)n * DIM + chunk * 4;
    atomicAdd(base + 0, (unsigned long long)(long long)llrintf(v.x * FIXSCALE));
    atomicAdd(base + 1, (unsigned long long)(long long)llrintf(v.y * FIXSCALE));
    atomicAdd(base + 2, (unsigned long long)(long long)llrintf(v.z * FIXSCALE));
    atomicAdd(base + 3, (unsigned long long)(long long)llrintf(v.w * FIXSCALE));
}

// ---------- phase 3: EMA update -> new codebook ----------
__global__ void k_cbnew(const float* __restrict__ ema, const float* __restrict__ counts,
                        const unsigned long long* __restrict__ dw,
                        const int* __restrict__ cnt_i, float* __restrict__ cbnew) {
    int gid = blockIdx.x * 256 + threadIdx.x;   // < NCODE*DIM
    int n = gid >> 9;
    float dwv = (float)((double)(long long)dw[gid] * INV_FIXSCALE);
    float e = 0.99f * ema[gid] + 0.01f * dwv;
    float c = 0.99f * counts[n] + 0.01f * (float)cnt_i[n];
    cbnew[gid] = e / c;
}

// ---------- phase 4: gather quantized + loss partials ----------
__global__ __launch_bounds__(256)
void k_quant(const float* __restrict__ x, const int* __restrict__ idx,
             const float* __restrict__ cbnew, float* __restrict__ out,
             unsigned long long* __restrict__ loss_acc) {
    __shared__ float red[256];
    int gid   = blockIdx.x * 256 + threadIdx.x; // < NTOK*DIM/4
    int token = gid >> 7;
    int chunk = gid & 127;
    int n = idx[token];
    float4 q = ((const float4*)cbnew)[(size_t)n * 128 + chunk];
    ((float4*)out)[gid] = q;
    float4 v = ((const float4*)x)[gid];
    float dx = v.x - q.x, dy = v.y - q.y, dz = v.z - q.z, dw_ = v.w - q.w;
    red[threadIdx.x] = 0.5f * (dx * dx + dy * dy + dz * dz + dw_ * dw_);
    __syncthreads();
    for (int off = 128; off > 0; off >>= 1) {
        if (threadIdx.x < off) red[threadIdx.x] += red[threadIdx.x + off];
        __syncthreads();
    }
    if (threadIdx.x == 0)
        atomicAdd(loss_acc, (unsigned long long)(long long)llrintf(red[0] * FIXSCALE));
}

__global__ void k_final(const unsigned long long* __restrict__ loss_acc,
                        float* __restrict__ out_loss) {
    double s = (double)(long long)(*loss_acc) * INV_FIXSCALE;
    *out_loss = (float)(s / (double)QELEMS);
}

extern "C" void kernel_launch(void* const* d_in, const int* in_sizes, int n_in,
                              void* d_out, int out_size, void* d_ws, size_t ws_size,
                              hipStream_t stream) {
    (void)in_sizes; (void)n_in; (void)out_size; (void)ws_size;
    const float* x      = (const float*)d_in[0];   // [8,2048,512]
    const float* cb     = (const float*)d_in[1];   // [1024,512]
    const float* ema    = (const float*)d_in[2];   // [1024,512]
    const float* counts = (const float*)d_in[3];   // [1024]

    char* ws = (char*)d_ws;
    unsigned long long* dw   = (unsigned long long*)(ws + WS_DW);
    unsigned long long* loss = (unsigned long long*)(ws + WS_LOSS);
    int*    cnt_i = (int*)(ws + WS_CNT);
    int*    idx_i = (int*)(ws + WS_IDX);
    float*  csq   = (float*)(ws + WS_CSQ);
    float*  cbnew = (float*)(ws + WS_CBNEW);
    __bf16* cbhi  = (__bf16*)(ws + WS_CBHI);
    __bf16* cblo  = (__bf16*)(ws + WS_CBLO);

    float* q_out    = (float*)d_out;               // [NTOK*DIM]
    float* loss_out = q_out + QELEMS;              // [1]
    float* idxf_out = q_out + QELEMS + 1;          // [NTOK] (indices as float values)

    hipMemsetAsync(d_ws, 0, WS_ZERO_BYTES, stream);

    k_prep_cb<<<NCODE * DIM / 256, 256, 0, stream>>>(cb, cbhi, cblo);
    k_csq<<<NCODE / 256, 256, 0, stream>>>(cb, csq);
    k_argmin<<<NTOK / TPB_TOK, 64, TPB_TOK * DIM * 2 * sizeof(__bf16), stream>>>(
        x, cbhi, cblo, csq, idx_i, idxf_out);
    k_counts<<<NTOK / 256, 256, 0, stream>>>(idx_i, cnt_i);
    k_scatter<<<NTOK * DIM / 4 / 256, 256, 0, stream>>>(x, idx_i, dw);
    k_cbnew<<<NCODE * DIM / 256, 256, 0, stream>>>(ema, counts, dw, cnt_i, cbnew);
    k_quant<<<NTOK * DIM / 4 / 256, 256, 0, stream>>>(x, idx_i, cbnew, q_out, loss);
    k_final<<<1, 1, 0, stream>>>(loss, loss_out);
}
